// SimTransformer_26809185861683
// MI455X (gfx1250) — matrix operand order repaired and checked by reading, not compiled, not run on hardware
//
#include <hip/hip_runtime.h>

typedef __attribute__((ext_vector_type(16))) __bf16 v16bf;
typedef __attribute__((ext_vector_type(8)))  __bf16 v8bf;
typedef __attribute__((ext_vector_type(8)))  float  v8f;

#define B_ 4
#define N_ 4096
#define D_ 512
#define EPS_ATTN 0.1f
#define LN_EPS   1e-5f
#define NORM_EPS 1e-12f

// lds_S row stride (elements): 128 keys + 8 pad -> 272B row stride, avoids
// the 64-bank-aligned stride that would make 16-lane A-fragment reads conflict.
#define SLD 136

__device__ __forceinline__ v8f zero8() {
  v8f z;
#pragma unroll
  for (int i = 0; i < 8; ++i) z[i] = 0.0f;
  return z;
}

// A-operand (16x32 bf16): elements 0..7 <- K=16hi+{0..7}, 8..15 <- K=16hi+{8..15}
__device__ __forceinline__ v16bf cvtA_f32(const float* __restrict__ p0,
                                          const float* __restrict__ p1) {
  v16bf r;
#pragma unroll
  for (int i = 0; i < 8; ++i) { r[i] = (__bf16)p0[i]; r[8 + i] = (__bf16)p1[i]; }
  return r;
}

// B-operand (32x16 bf16): lane n+16h holds K=16h+{0..15} of column n (contiguous row)
__device__ __forceinline__ v16bf cvtB_f32(const float* __restrict__ p) {
  v16bf r;
#pragma unroll
  for (int i = 0; i < 16; ++i) r[i] = (__bf16)p[i];
  return r;
}

__device__ __forceinline__ v16bf joinA(v8bf lo, v8bf hi2) {
  v16bf r;
#pragma unroll
  for (int i = 0; i < 8; ++i) { r[i] = lo[i]; r[8 + i] = hi2[i]; }
  return r;
}

// ---------------------------------------------------------------------------
// Kernel 1: q = x @ Wqk^T  (bf16 WMMA, f32 acc) -> L2-normalize rows -> qn bf16
// ---------------------------------------------------------------------------
__global__ __launch_bounds__(256)
void proj_q_kernel(const float* __restrict__ x, const float* __restrict__ Wqk,
                   __bf16* __restrict__ qn) {
  const int b     = blockIdx.y;
  const int qbase = blockIdx.x * 16;
  const int tid   = threadIdx.x;
  const int w     = tid >> 5;
  const int l     = tid & 31;
  const int m     = l & 15;
  const int hi    = l >> 4;

  __shared__ float lds_q[16][512];
  __shared__ float red[16][16];

  v8f acc[4];
#pragma unroll
  for (int c = 0; c < 4; ++c) acc[c] = zero8();

  const float* xr = x + ((size_t)b * N_ + qbase + m) * D_;
  const int colbase = w * 64;

  for (int k = 0; k < 16; ++k) {
    const int kb = k * 32;
    v16bf A = cvtA_f32(xr + kb + 16 * hi, xr + kb + 16 * hi + 8);
#pragma unroll
    for (int c = 0; c < 4; ++c) {
      const float* wr = Wqk + (size_t)(colbase + c * 16 + m) * D_ + kb + 16 * hi;
      v16bf Bm = cvtB_f32(wr);
      acc[c] = __builtin_amdgcn_wmma_f32_16x16x32_bf16(false, A, false, Bm,
                                                       (short)0, acc[c], false, false);
    }
  }
#pragma unroll
  for (int c = 0; c < 4; ++c)
#pragma unroll
    for (int r = 0; r < 8; ++r)
      lds_q[r + 8 * hi][colbase + c * 16 + m] = acc[c][r];
  __syncthreads();

  const int rr = tid >> 4, cc = tid & 15;
  float s = 0.f;
#pragma unroll 4
  for (int i = 0; i < 32; ++i) { float v = lds_q[rr][cc * 32 + i]; s += v * v; }
  red[rr][cc] = s;
  __syncthreads();
  float n2 = 0.f;
#pragma unroll
  for (int i = 0; i < 16; ++i) n2 += red[rr][i];
  const float rn = 1.0f / fmaxf(sqrtf(n2), NORM_EPS);

  __bf16* orow = qn + ((size_t)b * N_ + qbase + rr) * D_ + cc * 32;
#pragma unroll 4
  for (int i = 0; i < 32; ++i) orow[i] = (__bf16)(lds_q[rr][cc * 32 + i] * rn);
}

// ---------------------------------------------------------------------------
// Kernel 2: v = x @ Wv^T  -> store transposed vT[b][e][n] bf16
// ---------------------------------------------------------------------------
__global__ __launch_bounds__(256)
void proj_v_kernel(const float* __restrict__ x, const float* __restrict__ Wv,
                   __bf16* __restrict__ vT) {
  const int b     = blockIdx.y;
  const int nbase = blockIdx.x * 16;
  const int tid   = threadIdx.x;
  const int w     = tid >> 5;
  const int l     = tid & 31;
  const int m     = l & 15;
  const int hi    = l >> 4;

  __shared__ float lds_v[16][513];   // +1 pad for conflict-free column reads

  v8f acc[4];
#pragma unroll
  for (int c = 0; c < 4; ++c) acc[c] = zero8();

  const float* xr = x + ((size_t)b * N_ + nbase + m) * D_;
  const int colbase = w * 64;

  for (int k = 0; k < 16; ++k) {
    const int kb = k * 32;
    v16bf A = cvtA_f32(xr + kb + 16 * hi, xr + kb + 16 * hi + 8);
#pragma unroll
    for (int c = 0; c < 4; ++c) {
      const float* wr = Wv + (size_t)(colbase + c * 16 + m) * D_ + kb + 16 * hi;
      v16bf Bm = cvtB_f32(wr);
      acc[c] = __builtin_amdgcn_wmma_f32_16x16x32_bf16(false, A, false, Bm,
                                                       (short)0, acc[c], false, false);
    }
  }
#pragma unroll
  for (int c = 0; c < 4; ++c)
#pragma unroll
    for (int r = 0; r < 8; ++r)
      lds_v[r + 8 * hi][colbase + c * 16 + m] = acc[c][r];
  __syncthreads();

  for (int e = tid; e < 512; e += 256) {
    v16bf pk;
#pragma unroll
    for (int r = 0; r < 16; ++r) pk[r] = (__bf16)lds_v[r][e];
    *(v16bf*)(vT + ((size_t)b * D_ + e) * N_ + nbase) = pk;
  }
}

// ---------------------------------------------------------------------------
// Kernel 3: fused thresholded cosine attention + LayerNorm
// grid (N/16, B), 256 threads (8 waves). Per iteration: 128 keys.
//  Phase 1: wave w computes S tile for keys [jj+16w, jj+16w+16) (16 WMMA),
//           thresholds, stores bf16 C-layout -> lds_S[16][128].
//  Phase 2: wave w owns out cols [64w, 64w+64): contracts S (16x128, 4 LDS
//           A-fragments) against vT (16 WMMA) into acc[4].
// Zero redundant FLOPs; queries preloaded in registers (16 A-fragments).
// ---------------------------------------------------------------------------
__global__ __launch_bounds__(256)
void attn_kernel(const __bf16* __restrict__ qn, const __bf16* __restrict__ vT,
                 const float* __restrict__ gamma, const float* __restrict__ beta,
                 float* __restrict__ out) {
  const int b     = blockIdx.y;
  const int qbase = blockIdx.x * 16;
  const int tid   = threadIdx.x;
  const int w     = tid >> 5;
  const int l     = tid & 31;
  const int m     = l & 15;
  const int hi    = l >> 4;

  __shared__ float  lds_O[16][512];
  __shared__ __bf16 lds_S[16][SLD];
  __shared__ float  red1[16][16];
  __shared__ float  red2[16][16];
  __shared__ float  mv[16][2];

  v8f acc[4];
#pragma unroll
  for (int c = 0; c < 4; ++c) acc[c] = zero8();

  // Preload all 16 query A-fragments (row M=m, K=512) into registers.
  const __bf16* qrow = qn + ((size_t)b * N_ + qbase + m) * D_;
  v16bf qA[16];
#pragma unroll
  for (int kk = 0; kk < 16; ++kk) {
    const int kb = kk * 32;
    v8bf a0 = *(const v8bf*)(qrow + kb + 16 * hi);
    v8bf a1 = *(const v8bf*)(qrow + kb + 16 * hi + 8);
    qA[kk] = joinA(a0, a1);
  }

  const int colbase = w * 64;

  for (int t = 0; t < 32; ++t) {
    const int jj = t * 128;

    // ---- Phase 1: S tile for keys [jj + 16w, jj + 16w + 16) ----
    const __bf16* krow = qn + ((size_t)b * N_ + jj + w * 16 + m) * D_;  // B col N=m
    if (t + 1 < 32) __builtin_prefetch(krow + (size_t)128 * D_, 0, 3);
    v8f s0 = zero8();
#pragma unroll
    for (int kk = 0; kk < 16; ++kk) {
      v16bf Bk = *(const v16bf*)(krow + kk * 32 + 16 * hi);
      s0 = __builtin_amdgcn_wmma_f32_16x16x32_bf16(false, qA[kk], false, Bk,
                                                   (short)0, s0, false, false);
    }
    // threshold (>0.1 else 0) and publish bf16 S tile (C layout: col n=m, rows r+8hi)
#pragma unroll
    for (int r = 0; r < 8; ++r) {
      float v = s0[r];
      lds_S[r + 8 * hi][w * 16 + m] = (__bf16)(v > EPS_ATTN ? v : 0.0f);
    }
    __syncthreads();

    // ---- Phase 2: acc += S[16x128] @ V[128 x 64(w)] ----
    const __bf16* vcol = vT + ((size_t)b * D_ + colbase + m) * N_ + jj + 16 * hi;
    if (t + 1 < 32) __builtin_prefetch(vcol + 128, 0, 3);
#pragma unroll
    for (int kt = 0; kt < 4; ++kt) {
      const __bf16* sr = &lds_S[m][kt * 32];
      v8bf p0 = *(const v8bf*)(sr + 16 * hi);
      v8bf p1 = *(const v8bf*)(sr + 16 * hi + 8);
      v16bf SA = joinA(p0, p1);
#pragma unroll
      for (int c = 0; c < 4; ++c) {
        v16bf VB = *(const v16bf*)(vcol + kt * 32 + (size_t)c * 16 * N_);
        acc[c] = __builtin_amdgcn_wmma_f32_16x16x32_bf16(false, SA, false, VB,
                                                         (short)0, acc[c], false, false);
      }
    }
    __syncthreads();   // protect lds_S before next iteration's stores
  }

  // ---- stage O (each wave owns its columns; plain stores) ----
#pragma unroll
  for (int c = 0; c < 4; ++c)
#pragma unroll
    for (int r = 0; r < 8; ++r)
      lds_O[r + 8 * hi][colbase + c * 16 + m] = acc[c][r];
  __syncthreads();

  // ---- fused LayerNorm over 512 cols per row ----
  const int rr = tid >> 4, cc = tid & 15;
  float s = 0.f, q2 = 0.f;
#pragma unroll 4
  for (int i = 0; i < 32; ++i) {
    float v = lds_O[rr][cc * 32 + i];
    s += v; q2 += v * v;
  }
  red1[rr][cc] = s;
  red2[rr][cc] = q2;
  __syncthreads();
  if (cc == 0) {
    float S = 0.f, Q = 0.f;
#pragma unroll
    for (int i = 0; i < 16; ++i) { S += red1[rr][i]; Q += red2[rr][i]; }
    const float mean = S * (1.0f / 512.0f);
    const float var  = Q * (1.0f / 512.0f) - mean * mean;
    mv[rr][0] = mean;
    mv[rr][1] = rsqrtf(var + LN_EPS);
  }
  __syncthreads();
  const float mean = mv[rr][0], rs = mv[rr][1];
  float* orow = out + ((size_t)b * N_ + qbase + rr) * D_ + cc * 32;
#pragma unroll 4
  for (int i = 0; i < 32; ++i) {
    const int c = cc * 32 + i;
    orow[i] = (lds_O[rr][c] - mean) * rs * gamma[c] + beta[c];
  }
}

// ---------------------------------------------------------------------------
extern "C" void kernel_launch(void* const* d_in, const int* in_sizes, int n_in,
                              void* d_out, int out_size, void* d_ws, size_t ws_size,
                              hipStream_t stream) {
  const float* x     = (const float*)d_in[0];
  const float* Wqk   = (const float*)d_in[1];
  const float* Wv    = (const float*)d_in[2];
  const float* gamma = (const float*)d_in[3];
  const float* beta  = (const float*)d_in[4];
  float* out = (float*)d_out;

  __bf16* qn = (__bf16*)d_ws;                 // [B][N][D] bf16
  __bf16* vT = qn + (size_t)B_ * N_ * D_;     // [B][D][N] bf16

  dim3 grid(N_ / 16, B_), blk(256);
  proj_q_kernel<<<grid, blk, 0, stream>>>(x, Wqk, qn);
  proj_v_kernel<<<grid, blk, 0, stream>>>(x, Wv, vT);
  attn_kernel<<<grid, blk, 0, stream>>>(qn, vT, gamma, beta, out);
}
